// EfficientUpdateFormerV3_39273180955221
// MI455X (gfx1250) — matrix operand order repaired and checked by reading, not compiled, not run on hardware
//
#include <hip/hip_runtime.h>

// ---------------------------------------------------------------------------
// EfficientUpdateFormerV3 forward for MI455X (gfx1250, wave32, WMMA).
// - All dense linears: f16-operand / f32-accumulate WMMA GEMM
//   (v_wmma_f32_16x16x32_f16), 256x128x32 tiles, 64x64 wave tiles
//   (16 WMMA : 16 ds_load_b128 per wave per K-step), double-buffered LDS,
//   one barrier per K-step.
// - Weights pre-transposed/converted to f16 [N][Kp] once per call (L2-resident).
// - Activations flow in f16 between GEMMs (LN / attention / GELU emit f16).
// - Attention einsums (~3% of FLOPs) in a VALU kernel with LDS tiling.
// Workspace: ~1.08 GB.
// ---------------------------------------------------------------------------

typedef _Float16 f16;
typedef __attribute__((ext_vector_type(8)))  _Float16 v8h;
typedef __attribute__((ext_vector_type(16))) _Float16 v16h;
typedef __attribute__((ext_vector_type(8)))  float    v8f;

#define IN_DIM 1110
#define HID    384
#define HID2   768
#define MLPH   1536
#define HEADS  8
#define HD     48
#define TSEQ   48
#define NPT    1024
#define NVIRT  64
#define NTOT   1088
#define NB     2
#define ATT_SCALE 0.14433756729740643f  // 48^-0.5

// ---------------------------- WMMA GEMM ------------------------------------
#define BMT 256
#define BNT 128
#define BKT 32
#define LDT 40  // padded LDS row stride (f16 elems): 80B rows, 16B aligned, conflict-free

__device__ __forceinline__ float gelu_tanh(float x) {
  float u = 0.7978845608028654f * (x + 0.044715f * x * x * x);
  return 0.5f * x * (1.0f + tanhf(u));
}

// C[M,N] = act(A[M,K] @ W[K,N] + bias) (+ resid).
// A: f16 (K multiple of 32, unguarded vector loads) or f32 (K even; float2 tail).
// Wt: f16, pre-transposed [N][Kp], Kp = round_up(K,32), zero padded.
template <bool AF16, bool OUTF16>
__global__ __launch_bounds__(256) void wmma_gemm_kernel(
    const void* __restrict__ Av, const f16* __restrict__ Wt,
    const float* __restrict__ bias, const float* resid,
    void* Cv, int M, int N, int K, int Kp, int act)
{
  __shared__ f16 As[2][BMT * LDT];  // 40 KB
  __shared__ f16 Bs[2][BNT * LDT];  // 20 KB; stored [n][k]: frag reads contiguous

  const int tid = threadIdx.x;
  const int m0  = blockIdx.y * BMT;
  const int n0  = blockIdx.x * BNT;

  // staging: A: 1 thread/row (32 f16 = 64B); B: thread t -> column (t&127), K-half (t>>7)
  const int aRow = tid;
  const int bN   = tid & 127;
  const int bK0  = (tid >> 7) * 16;

  const int wid  = tid >> 5;
  const int lane = tid & 31;
  const int wm = (wid & 3) * 64;   // wave tile: 64 rows x 64 cols
  const int wn = (wid >> 2) * 64;
  const int lr = lane & 15;
  const int lh = lane >> 4;

  v8f acc[4][4];
#pragma unroll
  for (int mi = 0; mi < 4; ++mi)
#pragma unroll
    for (int ni = 0; ni < 4; ++ni)
#pragma unroll
      for (int j = 0; j < 8; ++j) acc[mi][ni][j] = 0.f;

  const bool aOK = (m0 + aRow) < M;
  const bool bOK = (n0 + bN) < N;
  const f16*   Ah = (const f16*)Av;
  const float* Af = (const float*)Av;

  v8h ar[4], br0, br1;      // f16 staging regs (A: 32 f16/row)
  float ra[32];             // f32-A staging regs (only used when !AF16)
#pragma unroll
  for (int q = 0; q < 4; ++q)
#pragma unroll
    for (int i = 0; i < 8; ++i) ar[q][i] = (f16)0.f;

  const int KT = (K + BKT - 1) / BKT;

  auto fetch = [&](int kt) {
    const int k0 = kt * BKT;
    if constexpr (AF16) {
      if (aOK) {
        const f16* p = Ah + (size_t)(m0 + aRow) * K + k0;
#pragma unroll
        for (int q = 0; q < 4; ++q) ar[q] = *(const v8h*)(p + q * 8);
      }
    } else {
      // K is even for the f32 path (input projection, K=1110): float2 granularity
      const float* p = Af + (size_t)(m0 + aRow) * K + k0;
      if (aOK && (k0 + BKT) <= K) {      // fast path
#pragma unroll
        for (int i = 0; i < 16; ++i) {
          float2 t = *(const float2*)(p + 2 * i);
          ra[2 * i] = t.x; ra[2 * i + 1] = t.y;
        }
      } else {                            // guarded tail tile (float2 granular)
#pragma unroll
        for (int i = 0; i < 16; ++i) {
          int k = k0 + 2 * i;
          if (aOK && (k + 2) <= K) {
            float2 t = *(const float2*)(p + 2 * i);
            ra[2 * i] = t.x; ra[2 * i + 1] = t.y;
          } else {
            ra[2 * i] = 0.f; ra[2 * i + 1] = 0.f;
          }
        }
      }
    }
    if (bOK) {
      const f16* p = Wt + (size_t)(n0 + bN) * Kp + k0 + bK0;  // zero-padded, no guard
      br0 = *(const v8h*)p;
      br1 = *(const v8h*)(p + 8);
    } else {
#pragma unroll
      for (int i = 0; i < 8; ++i) { br0[i] = (f16)0.f; br1[i] = (f16)0.f; }
    }
  };

  auto commit = [&](int b) {
    f16* a = &As[b][aRow * LDT];
    if constexpr (AF16) {
#pragma unroll
      for (int q = 0; q < 4; ++q) *(v8h*)(a + q * 8) = ar[q];
    } else {
#pragma unroll
      for (int i = 0; i < 32; ++i) a[i] = (f16)ra[i];
    }
    f16* bp = &Bs[b][bN * LDT + bK0];
    *(v8h*)bp = br0;
    *(v8h*)(bp + 8) = br1;
  };

  auto mma_step = [&](int b) {
    v16h afr[4];
#pragma unroll
    for (int mi = 0; mi < 4; ++mi) {
      // A 16x32 frag: lane<16 -> row lr, K {0..7,16..23}; lane>=16 -> K {8..15,24..31}
      int base = (wm + mi * 16 + lr) * LDT + lh * 8;
      v8h a0 = *(const v8h*)&As[b][base];
      v8h a1 = *(const v8h*)&As[b][base + 16];
      afr[mi] = __builtin_shufflevector(a0, a1,
                  0, 1, 2, 3, 4, 5, 6, 7, 8, 9, 10, 11, 12, 13, 14, 15);
    }
#pragma unroll
    for (int ni = 0; ni < 4; ++ni) {
      // B 32x16 frag: lane<16 -> col lr, K {0..7,16..23}; lane>=16 -> K {8..15,24..31}
      int base = (wn + ni * 16 + lr) * LDT + lh * 8;
      v8h b0 = *(const v8h*)&Bs[b][base];
      v8h b1 = *(const v8h*)&Bs[b][base + 16];
      v16h bfr = __builtin_shufflevector(b0, b1,
                  0, 1, 2, 3, 4, 5, 6, 7, 8, 9, 10, 11, 12, 13, 14, 15);
#pragma unroll
      for (int mi = 0; mi < 4; ++mi) {
        acc[mi][ni] = __builtin_amdgcn_wmma_f32_16x16x32_f16(
            false, afr[mi], false, bfr, (short)0, acc[mi][ni], false, false);
      }
    }
  };

  // double-buffered pipeline: one barrier per K-step
  fetch(0);
  commit(0);
  __syncthreads();
  for (int kt = 1; kt < KT; ++kt) {
    fetch(kt);
    mma_step((kt - 1) & 1);
    commit(kt & 1);       // opposite buffer from the one being read
    __syncthreads();      // covers RAW on buf[kt&1] and WAR on buf[(kt+1)&1]
  }
  mma_step((KT - 1) & 1);

  // epilogue: C layout: VGPR j -> M = j + 8*(lane>=16); N = lane&15
#pragma unroll
  for (int ni = 0; ni < 4; ++ni) {
    int col = n0 + wn + ni * 16 + lr;
    if (col >= N) continue;
    float bv = bias ? bias[col] : 0.f;
#pragma unroll
    for (int mi = 0; mi < 4; ++mi) {
#pragma unroll
      for (int j = 0; j < 8; ++j) {
        int row = m0 + wm + mi * 16 + lh * 8 + j;
        if (row < M) {
          float v = acc[mi][ni][j] + bv;
          if (act) v = gelu_tanh(v);
          size_t idx = (size_t)row * N + col;
          if constexpr (OUTF16) {
            ((f16*)Cv)[idx] = (f16)v;
          } else {
            float* C = (float*)Cv;
            if (resid) v += resid[idx];
            C[idx] = v;
          }
        }
      }
    }
  }
}

// --------------------- weight transpose + f16 convert ----------------------
// Wt[n*Kp + k] = (f16)W[k*N + n], zero-padded for k in [K, Kp)
__global__ void wtrans_kernel(const float* __restrict__ W, f16* __restrict__ Wt,
                              int K, int N, int Kp)
{
  __shared__ float tile[32][33];
  const int n0 = blockIdx.x * 32, k0 = blockIdx.y * 32;
  const int tx = threadIdx.x, ty = threadIdx.y;
#pragma unroll
  for (int it = 0; it < 4; ++it) {
    int k = k0 + ty + 8 * it;
    tile[ty + 8 * it][tx] = (k < K) ? W[(size_t)k * N + n0 + tx] : 0.f;
  }
  __syncthreads();
#pragma unroll
  for (int it = 0; it < 4; ++it) {
    int n = ty + 8 * it;
    Wt[(size_t)(n0 + n) * Kp + k0 + tx] = (f16)tile[tx][n];
  }
}

// ---------------------------- Attention ------------------------------------
// Q:(S,n1,HID) f32; KV:(S,n2,2*HID) f32, k=[0:HID) v=[HID:2HID); O:(S,n1,HID) f16
__global__ __launch_bounds__(256) void attn_kernel(
    const float* __restrict__ Q, const float* __restrict__ KV,
    f16* __restrict__ O, int n1, int n2)
{
  extern __shared__ float smem[];
  float* sc  = smem;               // [16][1024] scores
  float* qs  = sc + 16 * 1024;     // [16][48]
  float* kvs = qs + 16 * HD;       // [32][48]

  const int s  = blockIdx.x / HEADS;
  const int h  = blockIdx.x % HEADS;
  const int q0 = blockIdx.y * 16;
  const int tid = threadIdx.x;

  const float* Qb = Q + (size_t)s * n1 * HID + h * HD;
  const float* Kb = KV + (size_t)s * n2 * HID2 + h * HD;
  const float* Vb = Kb + HID;

  for (int x = tid; x < 16 * HD; x += 256) {
    int r = x / HD, d = x - r * HD;
    qs[x] = Qb[(size_t)(q0 + r) * HID + d];
  }
  __syncthreads();

  const int jj = tid & 31;
  const int qg = tid >> 5;
  for (int j0 = 0; j0 < n2; j0 += 32) {
    for (int x = tid; x < 32 * HD; x += 256) {
      int r = x / HD, d = x - r * HD;
      kvs[x] = (j0 + r < n2) ? Kb[(size_t)(j0 + r) * HID2 + d] : 0.f;
    }
    __syncthreads();
    if (j0 + jj < n2) {
#pragma unroll
      for (int qq = qg; qq < 16; qq += 8) {
        float a = 0.f;
#pragma unroll
        for (int d = 0; d < HD; ++d) a += qs[qq * HD + d] * kvs[jj * HD + d];
        sc[qq * 1024 + j0 + jj] = a * ATT_SCALE;
      }
    }
    __syncthreads();
  }

  // softmax: one wave per 2 rows (wave32)
  const int wv = tid >> 5, lane = tid & 31;
  for (int qq = wv * 2; qq < wv * 2 + 2; ++qq) {
    float mx = -3.0e38f;
    for (int j = lane; j < n2; j += 32) mx = fmaxf(mx, sc[qq * 1024 + j]);
#pragma unroll
    for (int o = 16; o; o >>= 1) mx = fmaxf(mx, __shfl_xor(mx, o, 32));
    float sum = 0.f;
    for (int j = lane; j < n2; j += 32) {
      float e = __expf(sc[qq * 1024 + j] - mx);
      sc[qq * 1024 + j] = e;
      sum += e;
    }
#pragma unroll
    for (int o = 16; o; o >>= 1) sum += __shfl_xor(sum, o, 32);
    float inv = 1.0f / sum;
    for (int j = lane; j < n2; j += 32) sc[qq * 1024 + j] *= inv;
  }
  __syncthreads();

  // A @ V: 16x48 = 768 outputs = 256 threads * 3
  float oacc[3] = {0.f, 0.f, 0.f};
  for (int j0 = 0; j0 < n2; j0 += 32) {
    for (int x = tid; x < 32 * HD; x += 256) {
      int r = x / HD, d = x - r * HD;
      kvs[x] = (j0 + r < n2) ? Vb[(size_t)(j0 + r) * HID2 + d] : 0.f;
    }
    __syncthreads();
    int jmax = n2 - j0; if (jmax > 32) jmax = 32;
#pragma unroll
    for (int c = 0; c < 3; ++c) {
      int idx = tid * 3 + c;
      int qq = idx / HD, dd = idx - qq * HD;
      float a = 0.f;
      for (int j = 0; j < jmax; ++j) a += sc[qq * 1024 + j0 + j] * kvs[j * HD + dd];
      oacc[c] += a;
    }
    __syncthreads();
  }
#pragma unroll
  for (int c = 0; c < 3; ++c) {
    int idx = tid * 3 + c;
    int qq = idx / HD, dd = idx - qq * HD;
    O[(size_t)(s * n1 + q0 + qq) * HID + h * HD + dd] = (f16)oacc[c];
  }
}

// ---------------------------- LayerNorm (f32 -> f16) ------------------------
__global__ __launch_bounds__(256) void ln_kernel(
    const float* __restrict__ x, f16* __restrict__ y,
    const float* __restrict__ g, const float* __restrict__ b,
    int rows, float eps)
{
  const int wv = threadIdx.x >> 5;
  const int lane = threadIdx.x & 31;
  const int row = blockIdx.x * 8 + wv;
  if (row >= rows) return;
  const float* xr = x + (size_t)row * HID;
  float v[12];
  float s = 0.f;
#pragma unroll
  for (int i = 0; i < 12; ++i) { v[i] = xr[lane + 32 * i]; s += v[i]; }
#pragma unroll
  for (int o = 16; o; o >>= 1) s += __shfl_xor(s, o, 32);
  float m = s * (1.0f / HID);
  float vs = 0.f;
#pragma unroll
  for (int i = 0; i < 12; ++i) { float d = v[i] - m; vs += d * d; }
#pragma unroll
  for (int o = 16; o; o >>= 1) vs += __shfl_xor(vs, o, 32);
  float inv = rsqrtf(vs * (1.0f / HID) + eps);
  f16* yr = y + (size_t)row * HID;
  if (g) {
#pragma unroll
    for (int i = 0; i < 12; ++i) { int c = lane + 32 * i; yr[c] = (f16)((v[i] - m) * inv * g[c] + b[c]); }
  } else {
#pragma unroll
    for (int i = 0; i < 12; ++i) { int c = lane + 32 * i; yr[c] = (f16)((v[i] - m) * inv); }
  }
}

// ------------------------ layout shuffle kernels ---------------------------
__global__ void assemble_tokens_kernel(const float* __restrict__ proj,
                                       const float* __restrict__ virt,
                                       float* __restrict__ tokens)
{
  const int row = blockIdx.x;            // (b, n, t)
  const int bb = row / (NTOT * TSEQ);
  const int rem = row - bb * (NTOT * TSEQ);
  const int n = rem / TSEQ;
  const int tt = rem - n * TSEQ;
  const float* src = (n < NPT)
      ? proj + ((size_t)(bb * NPT + n) * TSEQ + tt) * HID
      : virt + (size_t)(n - NPT) * HID;  // broadcast over b,t
  float* dst = tokens + (size_t)row * HID;
  for (int c = threadIdx.x; c < HID; c += blockDim.x) dst[c] = src[c];
}

__global__ void split_spatial_kernel(const float* __restrict__ tokens,
                                     float* __restrict__ pt, float* __restrict__ vt)
{
  const int row = blockIdx.x;
  const int bb = row / (NTOT * TSEQ);
  const int rem = row - bb * (NTOT * TSEQ);
  const int n = rem / TSEQ;
  const int tt = rem - n * TSEQ;
  const int s = bb * TSEQ + tt;
  float* dst = (n < NPT)
      ? pt + ((size_t)s * NPT + n) * HID
      : vt + ((size_t)s * NVIRT + (n - NPT)) * HID;
  const float* src = tokens + (size_t)row * HID;
  for (int c = threadIdx.x; c < HID; c += blockDim.x) dst[c] = src[c];
}

__global__ void merge_spatial_kernel(const float* __restrict__ pt,
                                     const float* __restrict__ vt,
                                     float* __restrict__ tokens)
{
  const int row = blockIdx.x;
  const int bb = row / (NTOT * TSEQ);
  const int rem = row - bb * (NTOT * TSEQ);
  const int n = rem / TSEQ;
  const int tt = rem - n * TSEQ;
  const int s = bb * TSEQ + tt;
  const float* src = (n < NPT)
      ? pt + ((size_t)s * NPT + n) * HID
      : vt + ((size_t)s * NVIRT + (n - NPT)) * HID;
  float* dst = tokens + (size_t)row * HID;
  for (int c = threadIdx.x; c < HID; c += blockDim.x) dst[c] = src[c];
}

// ------------------------------- heads -------------------------------------
__global__ __launch_bounds__(256) void head_kernel(
    const float* __restrict__ tokens,
    const float* __restrict__ wf, const float* __restrict__ bfl,
    const float* __restrict__ wv, const float* __restrict__ bvc,
    float* __restrict__ out_fc, float* __restrict__ out_tok)
{
  const int wvid = threadIdx.x >> 5;
  const int lane = threadIdx.x & 31;
  const int row = blockIdx.x * 8 + wvid;
  const int total = NB * NPT * TSEQ;
  if (row >= total) return;
  const int bb = row / (NPT * TSEQ);
  const int rem = row - bb * (NPT * TSEQ);
  const int n = rem / TSEQ;
  const int tt = rem - n * TSEQ;
  const float* tr = tokens + ((size_t)(bb * NTOT + n) * TSEQ + tt) * HID;
  float* orow = out_tok + (size_t)row * HID;
  float a0 = 0, a1 = 0, a2 = 0, a3 = 0;
#pragma unroll
  for (int i = 0; i < 12; ++i) {
    int c = lane + 32 * i;
    float xv = tr[c];
    orow[c] = xv;
    a0 += xv * wf[c * 2 + 0];
    a1 += xv * wf[c * 2 + 1];
    a2 += xv * wv[c * 2 + 0];
    a3 += xv * wv[c * 2 + 1];
  }
#pragma unroll
  for (int o = 16; o; o >>= 1) {
    a0 += __shfl_xor(a0, o, 32); a1 += __shfl_xor(a1, o, 32);
    a2 += __shfl_xor(a2, o, 32); a3 += __shfl_xor(a3, o, 32);
  }
  if (lane == 0) {
    float* o = out_fc + (size_t)row * 4;
    o[0] = a0 + bfl[0]; o[1] = a1 + bfl[1];
    o[2] = a2 + bvc[0]; o[3] = a3 + bvc[1];
  }
}

// ----------------------------- host side -----------------------------------
struct BlkW {
  const f16 *wq, *wkv, *wo, *w1, *w2;
  const float *bq, *bkv, *bo, *b1, *b2;
  const float *cg, *cb;   // cross-block ctx affine (nullptr for self blocks)
};
struct WS { f16* sX; float* sQ; float* sKV; f16* sAO; f16* sH; };

static inline int roundup32(int k) { return (k + 31) & ~31; }

static inline void gemm_a32(hipStream_t st, const float* A, const f16* Wt,
                            const float* bias, float* C, int M, int N, int K)
{
  dim3 grid((N + BNT - 1) / BNT, (M + BMT - 1) / BMT);
  wmma_gemm_kernel<false, false><<<grid, 256, 0, st>>>(
      A, Wt, bias, nullptr, C, M, N, K, roundup32(K), 0);
}
static inline void gemm_h(hipStream_t st, const f16* A, const f16* Wt,
                          const float* bias, const float* resid, float* C,
                          int M, int N, int K)
{
  dim3 grid((N + BNT - 1) / BNT, (M + BMT - 1) / BMT);
  wmma_gemm_kernel<true, false><<<grid, 256, 0, st>>>(
      A, Wt, bias, resid, C, M, N, K, roundup32(K), 0);
}
static inline void gemm_hh(hipStream_t st, const f16* A, const f16* Wt,
                           const float* bias, f16* C, int M, int N, int K, int act)
{
  dim3 grid((N + BNT - 1) / BNT, (M + BMT - 1) / BMT);
  wmma_gemm_kernel<true, true><<<grid, 256, 0, st>>>(
      A, Wt, bias, nullptr, C, M, N, K, roundup32(K), act);
}

static inline void layernorm(hipStream_t st, const float* x, f16* y,
                             const float* g, const float* b, int rows, float eps)
{
  ln_kernel<<<(rows + 7) / 8, 256, 0, st>>>(x, y, g, b, rows, eps);
}

static const int ATTN_SHMEM = (16 * 1024 + 16 * HD + 32 * HD) * (int)sizeof(float);

static inline void attention(hipStream_t st, const float* Qp, const float* KVp,
                             f16* Op, int S, int n1, int n2)
{
  dim3 grid(S * HEADS, n1 / 16);
  attn_kernel<<<grid, 256, ATTN_SHMEM, st>>>(Qp, KVp, Op, n1, n2);
}

static inline void mlp_run(hipStream_t st, float* x, int rows, const BlkW& p, WS w)
{
  const int CH = 24576;  // caps sH at 24576*1536 f16
  for (int off = 0; off < rows; off += CH) {
    int m = rows - off; if (m > CH) m = CH;
    gemm_hh(st, w.sX + (size_t)off * HID, p.w1, p.b1, w.sH, m, MLPH, HID, 1);
    gemm_h(st, w.sH, p.w2, p.b2, x + (size_t)off * HID, x + (size_t)off * HID, m, HID, MLPH);
  }
}

static void self_block(hipStream_t st, float* x, int S, int n, const BlkW& p, WS w)
{
  int rows = S * n;
  layernorm(st, x, w.sX, nullptr, nullptr, rows, 1e-6f);
  gemm_h(st, w.sX, p.wq, p.bq, nullptr, w.sQ, rows, HID, HID);
  gemm_h(st, w.sX, p.wkv, p.bkv, nullptr, w.sKV, rows, HID2, HID);
  attention(st, w.sQ, w.sKV, w.sAO, S, n, n);
  gemm_h(st, w.sAO, p.wo, p.bo, x, x, rows, HID, HID);
  layernorm(st, x, w.sX, nullptr, nullptr, rows, 1e-6f);
  mlp_run(st, x, rows, p, w);
}

static void cross_block(hipStream_t st, float* x, const float* ctx,
                        int S, int n1, int n2, const BlkW& p, WS w)
{
  int r1 = S * n1, r2 = S * n2;
  layernorm(st, ctx, w.sX, p.cg, p.cb, r2, 1e-5f);     // ctxn (affine, eps=1e-5)
  gemm_h(st, w.sX, p.wkv, p.bkv, nullptr, w.sKV, r2, HID2, HID);
  layernorm(st, x, w.sX, nullptr, nullptr, r1, 1e-6f); // xn (sX reused after kv)
  gemm_h(st, w.sX, p.wq, p.bq, nullptr, w.sQ, r1, HID, HID);
  attention(st, w.sQ, w.sKV, w.sAO, S, n1, n2);
  gemm_h(st, w.sAO, p.wo, p.bo, x, x, r1, HID, HID);
  layernorm(st, x, w.sX, nullptr, nullptr, r1, 1e-6f);
  mlp_run(st, x, r1, p, w);
}

extern "C" void kernel_launch(void* const* d_in, const int* in_sizes, int n_in,
                              void* d_out, int out_size, void* d_ws, size_t ws_size,
                              hipStream_t stream)
{
  (void)in_sizes; (void)n_in; (void)out_size; (void)ws_size;
  // Input flattening (depth-first, dict insertion order):
  // 0:input 1:w_in 2:b_in 3:virtual
  // 4..33:   time[0..2]   (wq,bq,wkv,bkv,wo,bo,w1,b1,w2,b2) x3
  // 34..63:  sv[0..2]     (same 10) x3
  // 64..99:  v2p[0..2]    (attn6, mlp4, ctx_g, ctx_b) x3
  // 100..135:p2v[0..2]    (same 12) x3
  // 136:w_flow 137:b_flow 138:w_vc 139:b_vc
  const float* const* F = (const float* const*)d_in;
  const float* input  = F[0];
  const float* b_in   = F[2];
  const float* virt   = F[3];
  const float* w_flow = F[136]; const float* b_flow = F[137];
  const float* w_vc   = F[138]; const float* b_vc   = F[139];

  // ---- workspace carve; total ~1.08 GB ----
  char* base = (char*)d_ws;
  auto alloc = [&](size_t bytes) -> void* {
    void* r = (void*)base;
    base += (bytes + 255) & ~(size_t)255;
    return r;
  };
  const size_t MROWS = (size_t)NB * NTOT * TSEQ;   // 104448
  float* tokens = (float*)alloc(MROWS * HID * 4);
  float* pt     = (float*)alloc((size_t)NB * TSEQ * NPT * HID * 4);
  float* vt     = (float*)alloc((size_t)NB * TSEQ * NVIRT * HID * 4);
  WS w;
  w.sX  = (f16*)  alloc(MROWS * HID * 2);
  w.sQ  = (float*)alloc(MROWS * HID * 4);
  w.sKV = (float*)alloc(MROWS * HID2 * 4);
  w.sAO = (f16*)  alloc(MROWS * HID * 2);
  w.sH  = (f16*)  alloc((size_t)24576 * MLPH * 2);
  f16* wt = (f16*)alloc((size_t)22000000 * 2);     // f16 weight pool (~42 MB used)

  // ---- per-call weight convert + transpose to f16 [N][Kp] ----
  auto xform = [&](const float* W, int K, int N) -> const f16* {
    int Kp = roundup32(K);
    dim3 g(N / 32, Kp / 32);
    wtrans_kernel<<<g, dim3(32, 8), 0, stream>>>(W, wt, K, N, Kp);
    const f16* r = wt;
    wt += (size_t)N * Kp;
    return r;
  };
  auto mkblk = [&](int b, bool cross) -> BlkW {
    BlkW p;
    p.wq  = xform(F[b + 0], HID, HID);   p.bq  = F[b + 1];
    p.wkv = xform(F[b + 2], HID, HID2);  p.bkv = F[b + 3];
    p.wo  = xform(F[b + 4], HID, HID);   p.bo  = F[b + 5];
    p.w1  = xform(F[b + 6], HID, MLPH);  p.b1  = F[b + 7];
    p.w2  = xform(F[b + 8], MLPH, HID);  p.b2  = F[b + 9];
    p.cg  = cross ? F[b + 10] : nullptr;
    p.cb  = cross ? F[b + 11] : nullptr;
    return p;
  };
  const f16* w_in_t = xform(F[1], IN_DIM, HID);
  BlkW TW[3], SW[3], VW[3], PW[3];
  for (int i = 0; i < 3; ++i) {
    TW[i] = mkblk(4 + i * 10, false);
    SW[i] = mkblk(34 + i * 10, false);
    VW[i] = mkblk(64 + i * 12, true);
    PW[i] = mkblk(100 + i * 12, true);
  }

  const int rows_in = NB * NPT * TSEQ;  // 98304

  // input projection (f32 A path, K=1110) -> sKV (f32 scratch), then assemble
  gemm_a32(stream, input, w_in_t, b_in, w.sKV, rows_in, HID, IN_DIM);
  assemble_tokens_kernel<<<NB * NTOT * TSEQ, 128, 0, stream>>>(w.sKV, virt, tokens);

  for (int i = 0; i < 3; ++i) {
    // time attention: (b*ntot=2176) sequences of length 48 (contiguous view)
    self_block(stream, tokens, NB * NTOT, TSEQ, TW[i], w);
    // spatial view
    split_spatial_kernel<<<NB * NTOT * TSEQ, 128, 0, stream>>>(tokens, pt, vt);
    cross_block(stream, vt, pt, NB * TSEQ, NVIRT, NPT, VW[i], w);
    self_block(stream, vt, NB * TSEQ, NVIRT, SW[i], w);
    cross_block(stream, pt, vt, NB * TSEQ, NPT, NVIRT, PW[i], w);
    merge_spatial_kernel<<<NB * NTOT * TSEQ, 128, 0, stream>>>(pt, vt, tokens);
  }

  // heads + token copy: out = [flow|vis_conf] (B,1024,T,4) ++ tokens (B,1024,T,384)
  float* out_fc  = (float*)d_out;
  float* out_tok = out_fc + (size_t)rows_in * 4;
  head_kernel<<<rows_in / 8, 256, 0, stream>>>(tokens, w_flow, b_flow, w_vc, b_vc,
                                               out_fc, out_tok);
}
